// SequencePredictor_68977174773888
// MI455X (gfx1250) — matrix operand order repaired and checked by reading, not compiled, not run on hardware
//
#include <hip/hip_runtime.h>
#include <hip/hip_bf16.h>

// ---------------------------------------------------------------------------
// CDNA5 (gfx1250, wave32) seq2seq LSTM + vocab projection.
//   * Persistent 1-WG LSTM: c/h state in registers, [x,h] in LDS (A-frag
//     layout), f16 weights streamed from L2, v_wmma_f32_16x16x32_f16 GEMMs.
//   * Projection GEMM (2048x50000) in f16 WMMA w/ f32 accumulate; W_proj
//     packed f16 (25.6MB) stays resident in the 192MB L2.
// ---------------------------------------------------------------------------

typedef __attribute__((ext_vector_type(16))) _Float16 v16h;
typedef __attribute__((ext_vector_type(8)))  float    v8f;

#define B_    32
#define TX_   128
#define TY_   64
#define V_    50000
#define D_    128
#define H_    256
#define KLSTM (D_ + H_)        // 384
#define NLSTM (4 * H_)         // 1024
#define KT_L  (KLSTM / 32)     // 12 k-tiles per LSTM step
#define NT_L  (NLSTM / 16)     // 64 n-tiles per LSTM step
#define KT_P  (H_ / 32)        // 8  k-tiles for projection
#define NT_P  (V_ / 16)        // 3125 n-tiles for projection (exact!)
#define TSTEPS (TX_ + TY_)     // 192 sequential steps

__device__ __forceinline__ float sigm(float x) { return 1.0f / (1.0f + __expf(-x)); }

// ---------------------------------------------------------------------------
// Pack a row-major f32 (K,N) matrix into f16 WMMA B-fragments.
// Tile (kt,nt) = 512 halfs; lane L holds column nt*16+(L&15),
// K rows kt*32 + (j<8 ? 8h+j : 16+8h+(j-8)), h=L>>4, 16 halfs per lane
// (ISA 05_wmma B-matrix striping -> loads as 2x global_load_b128).
// ---------------------------------------------------------------------------
__global__ void pack_bfrag(const float* __restrict__ W, _Float16* __restrict__ P,
                           int K, int N, int KT, int NT) {
  long total = (long)KT * NT * 512;
  for (long o = (long)blockIdx.x * blockDim.x + threadIdx.x; o < total;
       o += (long)gridDim.x * blockDim.x) {
    int  pos  = (int)(o & 511);
    long tile = o >> 9;
    int  nt   = (int)(tile % NT);
    int  kt   = (int)(tile / NT);
    int  lane = pos >> 4, j = pos & 15;
    int  k    = kt * 32 + ((j & 8) << 1) + ((lane >> 4) << 3) + (j & 7);
    int  n    = nt * 16 + (lane & 15);
    float v   = (k < K && n < N) ? W[(long)k * N + n] : 0.0f;
    P[o] = (_Float16)v;
  }
}

// ---------------------------------------------------------------------------
// A-fragment (16x32 f16, ISA layout) scatter helpers:
//   element (m,k) -> lane = m + ((k%16)>=8 ? 16:0), half j = (k/16)*8 + (k%8)
// ---------------------------------------------------------------------------
__device__ __forceinline__ int afrag_lane(int m, int kin) {
  return m + (((kin & 15) >= 8) ? 16 : 0);
}
__device__ __forceinline__ int afrag_j(int kin) {
  return ((kin >> 4) << 3) + (kin & 7);
}

// ---------------------------------------------------------------------------
// Persistent single-workgroup LSTM (encoder then decoder).
// 512 threads = 16 waves. Wave w owns H-columns [w*16, w*16+16) for ALL four
// gates (ntiles w, w+16, w+32, w+48), so the full cell update is in-register.
// ---------------------------------------------------------------------------
__global__ __launch_bounds__(512)
void lstm_seq_kernel(const int* __restrict__ enc, const int* __restrict__ dec,
                     const int* __restrict__ len, const float* __restrict__ E,
                     const _Float16* __restrict__ Wenc_p, const float* __restrict__ benc,
                     const _Float16* __restrict__ Wdec_p, const float* __restrict__ bdec,
                     _Float16* __restrict__ hs_packed)
{
  __shared__ __align__(32) _Float16 xh[2 * KT_L * 512];   // 24 KB: [x|h] A-frags

  const int tid  = threadIdx.x;
  const int w    = tid >> 5;        // wave 0..15
  const int lane = tid & 31;
  const int n    = lane & 15;       // N within tile
  const int hi   = lane >> 4;       // C/D row offset select

  // ---- init: zero the h part (ktiles 4..11, both mtiles) ----
  for (int e = tid; e < 2 * 8 * 512; e += 512) {
    int mt = e / (8 * 512);
    int r  = e % (8 * 512);
    xh[(mt * KT_L + 4) * 512 + r] = (_Float16)0.0f;
  }
  // ---- build x fragments for t = 0 (embedding gather, f32 -> f16) ----
  for (int e = tid; e < B_ * D_; e += 512) {
    int b = e >> 7, d = e & 127;
    int tok = enc[b * TX_ + 0];
    float v = E[(long)tok * D_ + d];
    int ktile = d >> 5, kin = d & 31;
    xh[((b >> 4) * KT_L + ktile) * 512 + afrag_lane(b & 15, kin) * 16 + afrag_j(kin)] =
        (_Float16)v;
  }

  // lane-resident cell state: 16 (row,col) cells per lane, fixed across steps
  float c_reg[16], h_reg[16];
  int   len_r[16];
#pragma unroll
  for (int idx = 0; idx < 16; ++idx) {
    c_reg[idx] = 0.0f;
    h_reg[idx] = 0.0f;
    int mt = idx >> 3, r = idx & 7;
    len_r[idx] = len[mt * 16 + r + hi * 8];
  }

  __syncthreads();

  const v8f vzero = {};

  for (int t = 0; t < TSTEPS; ++t) {
    const bool is_dec = (t >= TX_);
    const _Float16* Wp   = is_dec ? Wdec_p : Wenc_p;
    const float*    bias = is_dec ? bdec   : benc;

    // ---- z = [x,h] @ W : 2 mtiles x 4 gates, K = 12 tiles of 32 ----
    v8f acc[2][4];
#pragma unroll
    for (int mt = 0; mt < 2; ++mt)
#pragma unroll
      for (int g = 0; g < 4; ++g) acc[mt][g] = vzero;

#pragma unroll
    for (int kt = 0; kt < KT_L; ++kt) {
      v16h a0 = *(const v16h*)&xh[(0 * KT_L + kt) * 512 + lane * 16];
      v16h a1 = *(const v16h*)&xh[(1 * KT_L + kt) * 512 + lane * 16];
#pragma unroll
      for (int g = 0; g < 4; ++g) {
        int  nt = g * 16 + w;
        v16h bb = *(const v16h*)&Wp[((long)(kt * NT_L + nt) << 9) + lane * 16];
        acc[0][g] = __builtin_amdgcn_wmma_f32_16x16x32_f16(
            false, a0, false, bb, (short)0, acc[0][g], false, false);
        acc[1][g] = __builtin_amdgcn_wmma_f32_16x16x32_f16(
            false, a1, false, bb, (short)0, acc[1][g], false, false);
      }
    }

    float bgate[4];
#pragma unroll
    for (int g = 0; g < 4; ++g) bgate[g] = bias[g * H_ + w * 16 + n];

    // ---- in-register cell update (TF BasicLSTMCell, forget bias 1.0) ----
#pragma unroll
    for (int idx = 0; idx < 16; ++idx) {
      int mt = idx >> 3, r = idx & 7;
      float iv = acc[mt][0][r] + bgate[0];
      float jv = acc[mt][1][r] + bgate[1];
      float fv = acc[mt][2][r] + bgate[2];
      float ov = acc[mt][3][r] + bgate[3];
      float cc = c_reg[idx];
      float nc = cc * sigm(fv + 1.0f) + sigm(iv) * tanhf(jv);
      float nh = tanhf(nc) * sigm(ov);
      bool upd = is_dec || (t < len_r[idx]);   // dynamic_rnn length masking
      c_reg[idx] = upd ? nc : cc;
      h_reg[idx] = upd ? nh : h_reg[idx];
    }

    __syncthreads();   // everyone finished reading xh for this step

    // ---- scatter new h into xh A-frags (and decoder hs to global) ----
    const int hcol = w * 16 + n;              // 0..255
    const int ktile_h = 4 + (hcol >> 5);
    const int kin_h   = hcol & 31;
    const int lane_h0 = (((kin_h & 15) >= 8) ? 16 : 0);
    const int j_h     = afrag_j(kin_h);
#pragma unroll
    for (int idx = 0; idx < 16; ++idx) {
      int mt  = idx >> 3, r = idx & 7;
      int row = mt * 16 + r + hi * 8;          // batch index b
      xh[(mt * KT_L + ktile_h) * 512 + (lane_h0 + (row & 15)) * 16 + j_h] =
          (_Float16)h_reg[idx];
      if (is_dec) {
        // hs A-frag row ri = b*TY + t  ->  matches out (B,TY,V) row order
        int ri  = row * TY_ + (t - TX_);
        int mtg = ri >> 4, mg = ri & 15;
        int ktg = hcol >> 5, king = hcol & 31;
        hs_packed[((long)(mtg * KT_P + ktg) << 9) + afrag_lane(mg, king) * 16 +
                  afrag_j(king)] = (_Float16)h_reg[idx];
      }
    }

    // ---- build x fragments for step t+1 ----
    if (t + 1 < TSTEPS) {
      int tn = t + 1;
      for (int e = tid; e < B_ * D_; e += 512) {
        int b = e >> 7, d = e & 127;
        int tok = (tn < TX_) ? enc[b * TX_ + tn] : dec[b * TY_ + (tn - TX_)];
        float v = E[(long)tok * D_ + d];
        int ktile = d >> 5, kin = d & 31;
        xh[((b >> 4) * KT_L + ktile) * 512 + afrag_lane(b & 15, kin) * 16 +
           afrag_j(kin)] = (_Float16)v;
      }
    }
    __syncthreads();
  }
}

// ---------------------------------------------------------------------------
// Projection: C(2048,50000) = hs(2048,256) @ W_proj(256,50000), f16 WMMA.
// Block = 256 threads (8 waves) -> 32x256 output tile; wave computes one
// mtile x 4 ntiles, 8 K-steps.  Output write (410MB) is the HBM bound.
// ---------------------------------------------------------------------------
__global__ __launch_bounds__(256)
void proj_gemm_kernel(const _Float16* __restrict__ A, const _Float16* __restrict__ Bp,
                      float* __restrict__ out)
{
  const int lane = threadIdx.x & 31;
  const int w    = threadIdx.x >> 5;                   // 0..7
  const int mt   = (blockIdx.y << 1) + (w & 1);        // global mtile 0..127
  const int ntg0 = blockIdx.x * 16 + ((w >> 1) << 2);  // first of 4 ntiles

  v8f acc[4];
#pragma unroll
  for (int g = 0; g < 4; ++g) acc[g] = (v8f){};

#pragma unroll
  for (int kt = 0; kt < KT_P; ++kt) {
    v16h a = *(const v16h*)&A[((long)(mt * KT_P + kt) << 9) + lane * 16];
    if (kt + 1 < KT_P)   // global_prefetch_b8 next K-slice of W_proj (L2 hit)
      __builtin_prefetch(&Bp[((long)((kt + 1) * (long)NT_P + ntg0) << 9) + lane * 16], 0, 1);
#pragma unroll
    for (int g = 0; g < 4; ++g) {
      int nt = ntg0 + g;
      if (nt < NT_P) {   // uniform per wave -> EXEC stays all-ones for WMMA
        v16h bb = *(const v16h*)&Bp[((long)(kt * (long)NT_P + nt) << 9) + lane * 16];
        acc[g] = __builtin_amdgcn_wmma_f32_16x16x32_f16(
            false, a, false, bb, (short)0, acc[g], false, false);
      }
    }
  }

  const int rowbase = mt * 16 + ((lane >> 4) << 3);
  const int col_in  = lane & 15;
#pragma unroll
  for (int g = 0; g < 4; ++g) {
    int nt = ntg0 + g;
    if (nt < NT_P) {
      long col = (long)nt * 16 + col_in;
#pragma unroll
      for (int r = 0; r < 8; ++r)
        out[(long)(rowbase + r) * V_ + col] = acc[g][r];
    }
  }
}

// ---------------------------------------------------------------------------
extern "C" void kernel_launch(void* const* d_in, const int* in_sizes, int n_in,
                              void* d_out, int out_size, void* d_ws, size_t ws_size,
                              hipStream_t stream)
{
  const int*   enc   = (const int*)d_in[0];
  const int*   dec   = (const int*)d_in[1];
  const int*   len   = (const int*)d_in[2];
  const float* E     = (const float*)d_in[3];
  const float* Wenc  = (const float*)d_in[4];
  const float* benc  = (const float*)d_in[5];
  const float* Wdec  = (const float*)d_in[6];
  const float* bdec  = (const float*)d_in[7];
  const float* Wproj = (const float*)d_in[8];
  float*       out   = (float*)d_out;
  (void)in_sizes; (void)n_in; (void)out_size; (void)ws_size;

  char*  ws  = (char*)d_ws;
  size_t off = 0;
  auto alloc = [&](size_t bytes) -> void* {
    void* p = ws + off;
    off += (bytes + 255) & ~(size_t)255;
    return p;
  };
  _Float16* Wenc_p  = (_Float16*)alloc((size_t)KT_L * NT_L * 512 * 2);  // 768 KB
  _Float16* Wdec_p  = (_Float16*)alloc((size_t)KT_L * NT_L * 512 * 2);  // 768 KB
  _Float16* Wproj_p = (_Float16*)alloc((size_t)KT_P * NT_P * 512 * 2);  // 25.6 MB
  _Float16* hs_p    = (_Float16*)alloc((size_t)128 * KT_P * 512 * 2);   // 1 MB

  pack_bfrag<<<1536, 256, 0, stream>>>(Wenc,  Wenc_p,  KLSTM, NLSTM, KT_L, NT_L);
  pack_bfrag<<<1536, 256, 0, stream>>>(Wdec,  Wdec_p,  KLSTM, NLSTM, KT_L, NT_L);
  pack_bfrag<<<50000, 256, 0, stream>>>(Wproj, Wproj_p, H_,    V_,    KT_P, NT_P);

  lstm_seq_kernel<<<1, 512, 0, stream>>>(enc, dec, len, E, Wenc_p, benc,
                                         Wdec_p, bdec, hs_p);

  proj_gemm_kernel<<<dim3((NT_P + 15) / 16, 64), 256, 0, stream>>>(hs_p, Wproj_p, out);
}
